// MultiHeadAttention_7627861918063
// MI455X (gfx1250) — matrix operand order repaired and checked by reading, not compiled, not run on hardware
//
#include <hip/hip_runtime.h>

// ---------------------------------------------------------------------------
// MI455X (gfx1250, wave32) multi-head attention, bf16 WMMA pipeline with
// async-to-LDS B-tile staging (double buffered) in the GEMMs.
//   x[2,2048,1024] f32; Wq/k/v/o [1024,1024] f32; b* [1024] f32 -> out f32.
// ---------------------------------------------------------------------------

typedef __attribute__((ext_vector_type(16))) __bf16 v16bf;
typedef __attribute__((ext_vector_type(8)))  __bf16 v8bf;
typedef __attribute__((ext_vector_type(8)))  float  v8f;

#define BATCH   2
#define SEQ     2048
#define HID     1024
#define HEADS   16
#define HDIM    64
#define MROWS   (BATCH * SEQ)          // 4096
#define BROWB   80                     // padded LDS row stride (64B data + 16B pad)

// D = A*B + C, bf16 inputs, f32 accumulate
__device__ __forceinline__ v8f wmma_bf16(v16bf a, v16bf b, v8f c) {
    return __builtin_amdgcn_wmma_f32_16x16x32_bf16(
        /*neg_a=*/false, a, /*neg_b=*/false, b,
        /*c_mod=*/(short)0, c, /*reuse_a=*/false, /*reuse_b=*/false);
}

// A-fragment (16x32, 16-bit): row = lane&15; lanes 0-15 hold k 0..15,
// lanes 16-31 hold k 16..31.  rowptr already includes the k-base.
__device__ __forceinline__ v16bf load_a_frag(const __bf16* rowptr, int lane) {
    const int khalf = (lane >> 4) & 1;
    v8bf lo = *reinterpret_cast<const v8bf*>(rowptr + khalf * 16);
    v8bf hi = *reinterpret_cast<const v8bf*>(rowptr + khalf * 16 + 8);
    v16bf a;
#pragma unroll
    for (int i = 0; i < 8; ++i) { a[i] = lo[i]; a[i + 8] = hi[i]; }
    return a;
}

__device__ __forceinline__ float rowmax16(float v) {
#pragma unroll
    for (int off = 1; off < 16; off <<= 1) v = fmaxf(v, __shfl_xor(v, off, 32));
    return v;
}
__device__ __forceinline__ float rowsum16(float v) {
#pragma unroll
    for (int off = 1; off < 16; off <<= 1) v += __shfl_xor(v, off, 32);
    return v;
}

// Per-lane async DMA of a 64(n) x 32(k) bf16 B tile into LDS (row stride 80B).
// 256 16-byte chunks; 128 threads issue 2 each.  Tracked by ASYNCcnt.
__device__ __forceinline__ void stage_b_tile_async(const __bf16* __restrict__ WT,
                                                   int bn, int k,
                                                   unsigned lds_base, int tid) {
#pragma unroll
    for (int r = 0; r < 2; ++r) {
        const int q  = tid + r * 128;          // chunk id 0..255
        const int n  = q >> 2;                 // 0..63
        const int kq = q & 3;                  // 16B sub-chunk within the row
        const unsigned loff = lds_base + (unsigned)(n * BROWB + kq * 16);
        const unsigned long long g =
            (unsigned long long)(uintptr_t)(WT + (size_t)(bn + n) * HID + k + kq * 8);
        asm volatile("global_load_async_to_lds_b128 %0, %1, off"
                     :: "v"(loff), "v"(g) : "memory");
    }
}

// ---------------------------------------------------------------------------
__global__ void cvt_f32_to_bf16(const float* __restrict__ in,
                                __bf16* __restrict__ out, int n) {
    int i = blockIdx.x * blockDim.x + threadIdx.x;
    if (i < n) out[i] = (__bf16)in[i];
}

// W[k][n] (f32) -> WT[n][k] (bf16): WMMA B-fragments become contiguous loads.
__global__ void cvt_transpose_bf16(const float* __restrict__ W,
                                   __bf16* __restrict__ WT) {
    int idx = blockIdx.x * blockDim.x + threadIdx.x;   // 1M threads
    int n = idx & (HID - 1);
    int k = idx >> 10;
    WT[(size_t)n * HID + k] = (__bf16)W[(size_t)k * HID + n];
}

// ---------------------------------------------------------------------------
// Y = A[4096,1024] @ W + bias.  Block (4 waves) computes 128x64; the 64x32 B
// tile is async-DMA'd into double-buffered LDS and shared by all 4 waves.
// Each wave: 2 M-subtiles x 4 N-tiles = 8 WMMAs per k-step; A fragments are
// register double-buffered so global loads overlap the matrix pipe.
// mode 0: bf16 [B,H,S,Dh] (Q,K)   mode 1: bf16 [B,H,Dh,S] (V^T)
// mode 2: f32  [M,1024]   (final projection -> d_out)
__global__ void gemm_wmma(const __bf16* __restrict__ A,
                          const __bf16* __restrict__ WT,
                          const float* __restrict__ bias,
                          void* __restrict__ outp, int mode) {
    __shared__ __attribute__((aligned(128))) char bufB[2][64 * BROWB];

    const int tid  = threadIdx.x;
    const int lane = tid & 31;
    const int wave = tid >> 5;
    const int bn = (blockIdx.x & 15) << 6;          // 16 n-blocks of 64
    const int m0 = ((blockIdx.x >> 4) << 7) + wave * 32;  // 32 m-blocks of 128
    const int nlo = lane & 15, hi = (lane >> 4) & 1;

    const char* bcur = &bufB[0][0];
    const char* bnxt = &bufB[1][0];

    const __bf16* arow0 = A + (size_t)(m0 + nlo) * HID;
    const __bf16* arow1 = arow0 + (size_t)16 * HID;

    // Prologue: stage B(k=0) into buffer 0, load A(k=0) fragments.
    stage_b_tile_async(WT, bn, 0, (unsigned)(uintptr_t)bcur, tid);
    v16bf a0 = load_a_frag(arow0, lane);
    v16bf a1 = load_a_frag(arow1, lane);
    asm volatile("s_wait_asynccnt 0x0" ::: "memory");
    __syncthreads();

    const v8f vzero = {};
    v8f acc[2][4];
#pragma unroll
    for (int i = 0; i < 2; ++i)
#pragma unroll
        for (int t = 0; t < 4; ++t) acc[i][t] = vzero;

    for (int k = 0; k < HID; k += 32) {
        const int kn = k + 32;
        if (kn < HID)                               // stage next B tile
            stage_b_tile_async(WT, bn, kn, (unsigned)(uintptr_t)bnxt, tid);

        v16bf na0 = a0, na1 = a1;
        if (kn < HID) {                             // next A fragments
            __builtin_prefetch(arow0 + kn + 32, 0, 0);
            na0 = load_a_frag(arow0 + kn, lane);
            na1 = load_a_frag(arow1 + kn, lane);
        }

        // B fragments from LDS: col n' = t*16 + nlo, k-rows 16*hi..+15.
#pragma unroll
        for (int t = 0; t < 4; ++t) {
            const v16bf b = *reinterpret_cast<const v16bf*>(
                bcur + (t * 16 + nlo) * BROWB + hi * 32);
            acc[0][t] = wmma_bf16(a0, b, acc[0][t]);
            acc[1][t] = wmma_bf16(a1, b, acc[1][t]);
        }

        asm volatile("s_wait_asynccnt 0x0" ::: "memory");
        __syncthreads();                            // publish next buffer
        a0 = na0; a1 = na1;
        const char* tmp = bcur; bcur = bnxt; bnxt = tmp;
    }

    // C/D layout: N = lane&15, M = v + 8*hi (per 16-row subtile).
#pragma unroll
    for (int i = 0; i < 2; ++i) {
        const int mbase = m0 + i * 16 + (hi << 3);
#pragma unroll
        for (int t = 0; t < 4; ++t) {
            const int n = bn + (t << 4) + nlo;
            const float bv = bias[n];
            const int h = n >> 6, d = n & (HDIM - 1);
#pragma unroll
            for (int v = 0; v < 8; ++v) {
                const int m = mbase + v;
                const float val = acc[i][t][v] + bv;
                const int b = m >> 11, s = m & (SEQ - 1);
                if (mode == 0) {
                    ((__bf16*)outp)[(((size_t)b * HEADS + h) * SEQ + s) * HDIM + d] =
                        (__bf16)val;
                } else if (mode == 1) {
                    ((__bf16*)outp)[(((size_t)b * HEADS + h) * HDIM + d) * SEQ + s] =
                        (__bf16)val;
                } else {
                    ((float*)outp)[(size_t)m * HID + n] = val;
                }
            }
        }
    }
}

// ---------------------------------------------------------------------------
// Flash-style attention. One wave owns 16 query rows of one (b,h); iterates
// keys in chunks of 32 with online softmax. P tile is re-staged through LDS
// to convert the C/D VGPR layout into the A-fragment layout for P@V.
__global__ void attn_wmma(const __bf16* __restrict__ Q,
                          const __bf16* __restrict__ K,
                          const __bf16* __restrict__ Vt,
                          __bf16* __restrict__ AO) {
    __shared__ __attribute__((aligned(128))) __bf16 pbuf[4][16 * 32];

    const int lane = threadIdx.x & 31;
    const int wave = threadIdx.x >> 5;
    const int tile = blockIdx.x * 4 + wave;        // 0..4095
    const int bh = tile >> 7;                      // / (SEQ/16)
    const int qtile = tile & 127;
    const int b = bh >> 4, h = bh & 15;
    const int nlo = lane & 15, hi = (lane >> 4) & 1;

    const __bf16* Qbh = Q  + (size_t)bh * SEQ * HDIM;
    const __bf16* Kbh = K  + (size_t)bh * SEQ * HDIM;
    const __bf16* Vbh = Vt + (size_t)bh * HDIM * SEQ;
    __bf16* pb = pbuf[wave];

    const __bf16* qrow = Qbh + (size_t)(qtile * 16 + nlo) * HDIM;
    const v16bf qa0 = load_a_frag(qrow, lane);
    const v16bf qa1 = load_a_frag(qrow + 32, lane);

    v8f o0{}, o1{}, o2{}, o3{};
    float mrow[8], lrow[8];
#pragma unroll
    for (int v = 0; v < 8; ++v) { mrow[v] = -1e30f; lrow[v] = 0.0f; }

    const float scale = 0.125f;                    // 1/sqrt(64)

    for (int kc = 0; kc < SEQ; kc += 32) {
        // K^T B-fragments: col = key, rows = head-dim (contiguous in K[key][*])
        const __bf16* kp0 = Kbh + (size_t)(kc + nlo) * HDIM + (hi << 4);
        const v16bf kb0a = *reinterpret_cast<const v16bf*>(kp0);
        const v16bf kb0b = *reinterpret_cast<const v16bf*>(kp0 + 32);
        const __bf16* kp1 = kp0 + 16 * HDIM;
        const v16bf kb1a = *reinterpret_cast<const v16bf*>(kp1);
        const v16bf kb1b = *reinterpret_cast<const v16bf*>(kp1 + 32);
        __builtin_prefetch(kp0 + 32 * HDIM, 0, 0); // next key chunk

        // V B-fragments are independent of the softmax -> issue early so the
        // load latency hides under the exp/shuffle VALU chain below.
        const __bf16* vp = Vbh + (size_t)nlo * SEQ + kc + (hi << 4);
        const v16bf vb0 = *reinterpret_cast<const v16bf*>(vp);
        const v16bf vb1 = *reinterpret_cast<const v16bf*>(vp + (size_t)16 * SEQ);
        const v16bf vb2 = *reinterpret_cast<const v16bf*>(vp + (size_t)32 * SEQ);
        const v16bf vb3 = *reinterpret_cast<const v16bf*>(vp + (size_t)48 * SEQ);

        v8f c0{}, c1{};
        c0 = wmma_bf16(qa0, kb0a, c0);
        c0 = wmma_bf16(qa1, kb0b, c0);
        c1 = wmma_bf16(qa0, kb1a, c1);
        c1 = wmma_bf16(qa1, kb1b, c1);

        // Online softmax per row M = v + 8*hi (one 16-lane group per row).
#pragma unroll
        for (int v = 0; v < 8; ++v) {
            const float s0 = c0[v] * scale;
            const float s1 = c1[v] * scale;
            const float mx = rowmax16(fmaxf(s0, s1));
            const float mnew = fmaxf(mrow[v], mx);
            const float corr = __expf(mrow[v] - mnew);
            const float p0 = __expf(s0 - mnew);
            const float p1 = __expf(s1 - mnew);
            lrow[v] = lrow[v] * corr + rowsum16(p0 + p1);
            mrow[v] = mnew;
            o0[v] *= corr; o1[v] *= corr; o2[v] *= corr; o3[v] *= corr;
            const int prow = v + (hi << 3);
            pb[prow * 32 + nlo]      = (__bf16)p0;
            pb[prow * 32 + 16 + nlo] = (__bf16)p1;
        }

        // Same-wave LDS store->load; make the ordering explicit.
        asm volatile("s_wait_dscnt 0x0" ::: "memory");

        // P (16x32) A-fragment from LDS, then P @ V chunk (32 x 64).
        const v16bf pa = load_a_frag(pb + nlo * 32, lane);
        o0 = wmma_bf16(pa, vb0, o0);
        o1 = wmma_bf16(pa, vb1, o1);
        o2 = wmma_bf16(pa, vb2, o2);
        o3 = wmma_bf16(pa, vb3, o3);
    }

    // Normalize and write attn output as bf16 [B, S, H*Dh].
#pragma unroll
    for (int v = 0; v < 8; ++v) {
        const float inv = 1.0f / lrow[v];
        const int s = qtile * 16 + v + (hi << 3);
        const size_t base = ((size_t)b * SEQ + s) * HID + h * HDIM + nlo;
        AO[base]      = (__bf16)(o0[v] * inv);
        AO[base + 16] = (__bf16)(o1[v] * inv);
        AO[base + 32] = (__bf16)(o2[v] * inv);
        AO[base + 48] = (__bf16)(o3[v] * inv);
    }
}

// ---------------------------------------------------------------------------
extern "C" void kernel_launch(void* const* d_in, const int* in_sizes, int n_in,
                              void* d_out, int out_size, void* d_ws, size_t ws_size,
                              hipStream_t stream) {
    const float* x  = (const float*)d_in[0];
    const float* Wq = (const float*)d_in[1];
    const float* bq = (const float*)d_in[2];
    const float* Wk = (const float*)d_in[3];
    const float* bk = (const float*)d_in[4];
    const float* Wv = (const float*)d_in[5];
    const float* bv = (const float*)d_in[6];
    const float* Wo = (const float*)d_in[7];
    const float* bo = (const float*)d_in[8];

    char* ws = (char*)d_ws;
    const size_t MB = 1024ull * 1024ull;
    __bf16* Xb  = (__bf16*)(ws);             // 8 MiB: x in bf16
    __bf16* WTq = (__bf16*)(ws + 8 * MB);    // 2 MiB each, transposed bf16
    __bf16* WTk = (__bf16*)(ws + 10 * MB);
    __bf16* WTv = (__bf16*)(ws + 12 * MB);
    __bf16* WTo = (__bf16*)(ws + 14 * MB);
    __bf16* Qb  = (__bf16*)(ws + 16 * MB);   // 8 MiB [B,H,S,Dh]
    __bf16* Kb  = (__bf16*)(ws + 24 * MB);   // 8 MiB [B,H,S,Dh]
    __bf16* Vt  = (__bf16*)(ws + 32 * MB);   // 8 MiB [B,H,Dh,S]
    __bf16* AO  = (__bf16*)(ws + 40 * MB);   // 8 MiB [B,S,H*Dh]

    const int NX = MROWS * HID;              // 4M elements
    cvt_f32_to_bf16<<<NX / 256, 256, 0, stream>>>(x, Xb, NX);
    cvt_transpose_bf16<<<(HID * HID) / 256, 256, 0, stream>>>(Wq, WTq);
    cvt_transpose_bf16<<<(HID * HID) / 256, 256, 0, stream>>>(Wk, WTk);
    cvt_transpose_bf16<<<(HID * HID) / 256, 256, 0, stream>>>(Wv, WTv);
    cvt_transpose_bf16<<<(HID * HID) / 256, 256, 0, stream>>>(Wo, WTo);

    // 512 blocks x 128 threads: each block owns a 128x64 output tile.
    gemm_wmma<<<512, 128, 0, stream>>>(Xb, WTq, bq, (void*)Qb, 0);
    gemm_wmma<<<512, 128, 0, stream>>>(Xb, WTk, bk, (void*)Kb, 0);
    gemm_wmma<<<512, 128, 0, stream>>>(Xb, WTv, bv, (void*)Vt, 1);

    attn_wmma<<<1024, 128, 0, stream>>>(Qb, Kb, Vt, AO);

    gemm_wmma<<<512, 128, 0, stream>>>(AO, WTo, bo, d_out, 2);
}
